// MultiHeadAttention_163208757552
// MI455X (gfx1250) — matrix operand order repaired and checked by reading, not compiled, not run on hardware
//
#include <hip/hip_runtime.h>

typedef __bf16 bf16;
typedef __attribute__((ext_vector_type(16))) __bf16 v16bf;
typedef __attribute__((ext_vector_type(8)))  __bf16 v8bf;
typedef __attribute__((ext_vector_type(4)))  __bf16 v4bf;
typedef __attribute__((ext_vector_type(8)))  float  v8f;
typedef __attribute__((ext_vector_type(4)))  unsigned int u32x4;
typedef __attribute__((ext_vector_type(4)))  int i32x4;
typedef __attribute__((ext_vector_type(8)))  int i32x8;

#define WMMA_BF16(a, b, c) \
  __builtin_amdgcn_wmma_f32_16x16x32_bf16(false, (a), false, (b), (short)0, (c), false, false)

// ---------------------------------------------------------------------------
// Fragment loaders (contiguous k order: lane half h holds k = 16h..16h+15, wave32)
// ---------------------------------------------------------------------------

// A fragment 16(M) x 32(K), bf16 row-major source, ld in elements.
__device__ __forceinline__ v16bf ld_a_bf16(const bf16* base, int ld, int lane) {
  const bf16* p = base + (size_t)(lane & 15) * ld + ((lane >> 4) << 4);
  v8bf lo = *reinterpret_cast<const v8bf*>(p);
  v8bf hi = *reinterpret_cast<const v8bf*>(p + 8);
  v16bf a;
#pragma unroll
  for (int i = 0; i < 8; ++i) { a[i] = lo[i]; a[i + 8] = hi[i]; }
  return a;
}

// B fragment 32(K) x 16(N) from a K-contiguous source: element (n,k) at n*ld + k.
__device__ __forceinline__ v16bf ld_b_kc(const bf16* base, int ld, int lane) {
  const bf16* p = base + (size_t)(lane & 15) * ld + ((lane >> 4) << 4);
  v8bf lo = *reinterpret_cast<const v8bf*>(p);
  v8bf hi = *reinterpret_cast<const v8bf*>(p + 8);
  v16bf b;
#pragma unroll
  for (int i = 0; i < 8; ++i) { b[i] = lo[i]; b[i + 8] = hi[i]; }
  return b;
}

// ---------------------------------------------------------------------------
// TDM: DMA one V tile [32 rows x 64 bf16] global -> LDS, padded rows (stride 68 el)
// D# per CDNA5 ISA 8.3/8.4: group0 = {flags, lds_addr, global_addr, type=2},
// group1 = {data_size=2B, pad 8B per 128B row, dims/strides}.
// 6-operand builtin on this toolchain: (g0 v4u, g1 v8i, g2 v4i, g3 v4i, v8i, cpol).
// ---------------------------------------------------------------------------
__device__ __forceinline__ void tdm_load_v_tile(const bf16* gsrc, unsigned lds_off) {
  unsigned long long ga = (unsigned long long)(uintptr_t)gsrc;
  u32x4 g0;
  g0[0] = 1u;                                              // count=1 (valid), user mode
  g0[1] = lds_off;                                         // lds_addr (bytes)
  g0[2] = (unsigned)ga;                                    // global_addr[31:0]
  g0[3] = ((unsigned)(ga >> 32) & 0x01FFFFFFu) | 0x80000000u; // ga[56:32] | type=2
  i32x8 g1;
  g1[0] = (1 << 16)      // data_size = 2 bytes
        | (1 << 20)      // pad_enable
        | (4 << 22)      // pad_interval: every 32 dwords (=128B row)
        | (1 << 25);     // pad_amount: 2 dwords (=8B)  -> row stride 136B = 68 el
  g1[1] = (64 << 16);    // tensor_dim0 = 64
  g1[2] = (2048 << 16);  // tensor_dim1 = 2048
  g1[3] = (64 << 16);    // tile_dim0 = 64
  g1[4] = 32;            // tile_dim1 = 32, tile_dim2 = 0
  g1[5] = 64;            // tensor_dim0_stride = 64
  g1[6] = 0;
  g1[7] = 0;
  i32x4 gz;
  gz[0] = gz[1] = gz[2] = gz[3] = 0;
  i32x8 gz8;
#pragma unroll
  for (int i = 0; i < 8; ++i) gz8[i] = 0;
  __builtin_amdgcn_tensor_load_to_lds(g0, g1, gz, gz, gz8, 0);
}

// ---------------------------------------------------------------------------
// Kernel: fp32 -> bf16 elementwise (x), 4 elements/thread
// ---------------------------------------------------------------------------
__global__ void __launch_bounds__(256) k_cvt4(const float* __restrict__ src,
                                              bf16* __restrict__ dst, int n4) {
  int idx = blockIdx.x * 256 + threadIdx.x;
  if (idx < n4) {
    float4 f = reinterpret_cast<const float4*>(src)[idx];
    v4bf o;
    o[0] = (bf16)f.x; o[1] = (bf16)f.y; o[2] = (bf16)f.z; o[3] = (bf16)f.w;
    reinterpret_cast<v4bf*>(dst)[idx] = o;
  }
}

// ---------------------------------------------------------------------------
// Kernel: fp32 [K][N] -> bf16 transposed [N][K]
// ---------------------------------------------------------------------------
__global__ void __launch_bounds__(256) k_cvt_transpose(const float* __restrict__ src,
                                                       bf16* __restrict__ dst,
                                                       int K, int N) {
  int idx = blockIdx.x * 256 + threadIdx.x;
  if (idx < K * N) {
    int k = idx / N, n = idx % N;
    dst[(size_t)n * K + k] = (bf16)src[idx];
  }
}

// ---------------------------------------------------------------------------
// Kernel 1: QKV GEMM  xb bf16 [8192][1024] @ WqkvT bf16 [3072][1024] + bias
// wave tile 16(M) x 64(N); explicit double-buffered k-loop (no rotation moves);
// scatter bf16 into Q/K/V [B,H,T,64]
// ---------------------------------------------------------------------------
__global__ void __launch_bounds__(256) k_qkv_gemm(const bf16* __restrict__ xb,
                                                  const bf16* __restrict__ WT,
                                                  const float* __restrict__ bias,
                                                  bf16* __restrict__ Qb,
                                                  bf16* __restrict__ Kb,
                                                  bf16* __restrict__ Vb) {
  int lane = threadIdx.x & 31;
  int wv = blockIdx.x * 8 + (threadIdx.x >> 5);
  int mt = wv / 48, nt = wv % 48;
  int m0 = mt * 16, n0 = nt * 64;

  v8f acc[4];
#pragma unroll
  for (int j = 0; j < 4; ++j)
#pragma unroll
    for (int r = 0; r < 8; ++r) acc[j][r] = 0.f;

  const bf16* ya = xb + (size_t)m0 * 1024;
  const bf16* wb[4];
#pragma unroll
  for (int j = 0; j < 4; ++j) wb[j] = WT + (size_t)(n0 + 16 * j) * 1024;

  v16bf a0 = ld_a_bf16(ya, 1024, lane);
  v16bf b0[4];
#pragma unroll
  for (int j = 0; j < 4; ++j) b0[j] = ld_b_kc(wb[j], 1024, lane);

#pragma unroll 1
  for (int kk = 0; kk < 1024; kk += 64) {
    v16bf a1 = ld_a_bf16(ya + kk + 32, 1024, lane);
    v16bf b1[4];
#pragma unroll
    for (int j = 0; j < 4; ++j) b1[j] = ld_b_kc(wb[j] + kk + 32, 1024, lane);
#pragma unroll
    for (int j = 0; j < 4; ++j) acc[j] = WMMA_BF16(a0, b0[j], acc[j]);
    if (kk + 64 < 1024) {
      a0 = ld_a_bf16(ya + kk + 64, 1024, lane);
#pragma unroll
      for (int j = 0; j < 4; ++j) b0[j] = ld_b_kc(wb[j] + kk + 64, 1024, lane);
    }
#pragma unroll
    for (int j = 0; j < 4; ++j) acc[j] = WMMA_BF16(a1, b1[j], acc[j]);
  }

  int mrb = (lane >> 4) << 3;
#pragma unroll
  for (int j = 0; j < 4; ++j) {
    int n = n0 + 16 * j + (lane & 15);
    float bv = bias[n];
    int sel = n >> 10;
    int nn = n & 1023;
    int h = nn >> 6, d = nn & 63;
    bf16* dstb = (sel == 0) ? Qb : ((sel == 1) ? Kb : Vb);
#pragma unroll
    for (int r = 0; r < 8; ++r) {
      int row = m0 + r + mrb;
      int bb = row >> 11, t = row & 2047;
      dstb[((((size_t)bb * 16 + h) * 2048) + t) * 64 + d] = (bf16)(acc[j][r] + bv);
    }
  }
}

// ---------------------------------------------------------------------------
// Kernel 2: causal flash attention, one 16-query tile per wave; V via TDM
// ---------------------------------------------------------------------------
#define VSTR 68   // LDS V row stride in elements (TDM pad: +8B per 128B row)

__global__ void __launch_bounds__(256) k_attn(const bf16* __restrict__ Qb,
                                              const bf16* __restrict__ Kb,
                                              const bf16* __restrict__ Vb,
                                              bf16* __restrict__ Y) {
  __shared__ __align__(16) bf16 smem[8 * (16 * 32 + 32 * VSTR)];
  int lane = threadIdx.x & 31;
  int warp = threadIdx.x >> 5;
  int wv = blockIdx.x * 8 + warp;
  int qt = wv & 127;
  int bh = wv >> 7;
  int q0 = qt * 16;

  bf16* Pl = smem + warp * (16 * 32 + 32 * VSTR);
  bf16* Vl = Pl + 16 * 32;
  unsigned vl_lds_off = (unsigned)(uintptr_t)(void*)Vl;   // low 32 bits = LDS byte offset

  const bf16* qbase = Qb + ((size_t)bh * 2048 + q0) * 64;
  const bf16* kbase = Kb + (size_t)bh * 2048 * 64;
  const bf16* vbase = Vb + (size_t)bh * 2048 * 64;

  v16bf aq0 = ld_a_bf16(qbase, 64, lane);
  v16bf aq1 = ld_a_bf16(qbase + 32, 64, lane);

  v8f o[4];
  float M[8], L[8];
#pragma unroll
  for (int r = 0; r < 8; ++r) {
    o[0][r] = o[1][r] = o[2][r] = o[3][r] = 0.f;
    M[r] = -1e30f; L[r] = 0.f;
  }
  int mrb = (lane >> 4) << 3;

  for (int kb = 0; kb <= q0 + 15; kb += 32) {
    // previous iteration's LDS reads must finish before TDM overwrites the V tile
    asm volatile("s_wait_dscnt 0x0" ::: "memory");
    tdm_load_v_tile(vbase + (size_t)kb * 64, vl_lds_off);

    // S = Q K^T (two 16x16 tiles over 32 keys), overlaps the TDM DMA
    v8f s0, s1;
#pragma unroll
    for (int r = 0; r < 8; ++r) { s0[r] = 0.f; s1[r] = 0.f; }
    {
      v16bf b00 = ld_b_kc(kbase + (size_t)kb * 64, 64, lane);
      v16bf b01 = ld_b_kc(kbase + (size_t)kb * 64 + 32, 64, lane);
      v16bf b10 = ld_b_kc(kbase + (size_t)(kb + 16) * 64, 64, lane);
      v16bf b11 = ld_b_kc(kbase + (size_t)(kb + 16) * 64 + 32, 64, lane);
      s0 = WMMA_BF16(aq0, b00, s0);
      s0 = WMMA_BF16(aq1, b01, s0);
      s1 = WMMA_BF16(aq0, b10, s1);
      s1 = WMMA_BF16(aq1, b11, s1);
    }

    bool need_mask = (kb + 31 > q0);
#pragma unroll
    for (int r = 0; r < 8; ++r) {
      s0[r] *= 0.125f;              // 1/sqrt(64)
      s1[r] *= 0.125f;
      if (need_mask) {
        int q = q0 + r + mrb;
        int k = kb + (lane & 15);
        if (k > q)      s0[r] = -1e30f;
        if (k + 16 > q) s1[r] = -1e30f;
      }
    }

    // online softmax: rows live across 16-lane halves -> xor reductions
#pragma unroll
    for (int r = 0; r < 8; ++r) {
      float t = fmaxf(s0[r], s1[r]);
      t = fmaxf(t, __shfl_xor(t, 1, 32));
      t = fmaxf(t, __shfl_xor(t, 2, 32));
      t = fmaxf(t, __shfl_xor(t, 4, 32));
      t = fmaxf(t, __shfl_xor(t, 8, 32));
      float mnew = fmaxf(M[r], t);
      float corr = __expf(M[r] - mnew);
      M[r] = mnew;
      float p0 = __expf(s0[r] - mnew);
      float p1 = __expf(s1[r] - mnew);
      s0[r] = p0; s1[r] = p1;
      float rs = p0 + p1;
      rs += __shfl_xor(rs, 1, 32);
      rs += __shfl_xor(rs, 2, 32);
      rs += __shfl_xor(rs, 4, 32);
      rs += __shfl_xor(rs, 8, 32);
      L[r] = L[r] * corr + rs;
      o[0][r] *= corr; o[1][r] *= corr; o[2][r] *= corr; o[3][r] *= corr;
    }

    // write P (16x32) to LDS row-major, reload as A fragment
    {
      int nloc = lane & 15;
#pragma unroll
      for (int r = 0; r < 8; ++r) {
        int m = r + mrb;
        Pl[m * 32 + nloc] = (bf16)s0[r];
        Pl[m * 32 + 16 + nloc] = (bf16)s1[r];
      }
    }
    asm volatile("s_wait_dscnt 0x0" ::: "memory");

    v16bf pf;
    {
      const bf16* p = Pl + (lane & 15) * 32 + ((lane >> 4) << 4);
      v8bf lo = *reinterpret_cast<const v8bf*>(p);
      v8bf hi = *reinterpret_cast<const v8bf*>(p + 8);
#pragma unroll
      for (int i = 0; i < 8; ++i) { pf[i] = lo[i]; pf[i + 8] = hi[i]; }
    }

    // V tile must be in LDS now
    __builtin_amdgcn_s_wait_tensorcnt(0);
    asm volatile("" ::: "memory");

    // O += P * V : gather V B-fragments (col d = 16*dt + lane%16, 16 consecutive keys)
    {
      const bf16* vp = Vl + (size_t)((lane >> 4) << 4) * VSTR + (lane & 15);
#pragma unroll
      for (int dt = 0; dt < 4; ++dt) {
        v16bf vf;
#pragma unroll
        for (int e = 0; e < 16; ++e) vf[e] = vp[(size_t)e * VSTR + dt * 16];
        o[dt] = WMMA_BF16(pf, vf, o[dt]);
      }
    }
  }

  // epilogue: normalize and store y[b, t, h*64 + d] as bf16
  int nloc = lane & 15;
  int b = bh >> 4, h = bh & 15;
#pragma unroll
  for (int r = 0; r < 8; ++r) {
    float rl = 1.0f / L[r];
    int row = b * 2048 + q0 + r + mrb;
    size_t base = (size_t)row * 1024 + h * 64 + nloc;
    Y[base + 0]  = (bf16)(o[0][r] * rl);
    Y[base + 16] = (bf16)(o[1][r] * rl);
    Y[base + 32] = (bf16)(o[2][r] * rl);
    Y[base + 48] = (bf16)(o[3][r] * rl);
  }
}

// ---------------------------------------------------------------------------
// Kernel 3: projection  Y bf16 [8192][1024] @ WprojT bf16 [1024][1024] + bias -> fp32
// ---------------------------------------------------------------------------
__global__ void __launch_bounds__(256) k_proj_gemm(const bf16* __restrict__ Y,
                                                   const bf16* __restrict__ WT,
                                                   const float* __restrict__ bias,
                                                   float* __restrict__ out) {
  int lane = threadIdx.x & 31;
  int wv = blockIdx.x * 8 + (threadIdx.x >> 5);
  int mt = wv >> 4, nt = wv & 15;
  int m0 = mt * 16, n0 = nt * 64;

  v8f acc[4];
#pragma unroll
  for (int j = 0; j < 4; ++j)
#pragma unroll
    for (int r = 0; r < 8; ++r) acc[j][r] = 0.f;

  const bf16* ya = Y + (size_t)m0 * 1024;
  const bf16* wb[4];
#pragma unroll
  for (int j = 0; j < 4; ++j) wb[j] = WT + (size_t)(n0 + 16 * j) * 1024;

  v16bf a0 = ld_a_bf16(ya, 1024, lane);
  v16bf b0[4];
#pragma unroll
  for (int j = 0; j < 4; ++j) b0[j] = ld_b_kc(wb[j], 1024, lane);

#pragma unroll 1
  for (int kk = 0; kk < 1024; kk += 64) {
    v16bf a1 = ld_a_bf16(ya + kk + 32, 1024, lane);
    v16bf b1[4];
#pragma unroll
    for (int j = 0; j < 4; ++j) b1[j] = ld_b_kc(wb[j] + kk + 32, 1024, lane);
#pragma unroll
    for (int j = 0; j < 4; ++j) acc[j] = WMMA_BF16(a0, b0[j], acc[j]);
    if (kk + 64 < 1024) {
      a0 = ld_a_bf16(ya + kk + 64, 1024, lane);
#pragma unroll
      for (int j = 0; j < 4; ++j) b0[j] = ld_b_kc(wb[j] + kk + 64, 1024, lane);
    }
#pragma unroll
    for (int j = 0; j < 4; ++j) acc[j] = WMMA_BF16(a1, b1[j], acc[j]);
  }

  int mrb = (lane >> 4) << 3;
#pragma unroll
  for (int j = 0; j < 4; ++j) {
    int n = n0 + 16 * j + (lane & 15);
    float bv = bias[n];
#pragma unroll
    for (int r = 0; r < 8; ++r) {
      int row = m0 + r + mrb;
      out[(size_t)row * 1024 + n] = acc[j][r] + bv;
    }
  }
}

// ---------------------------------------------------------------------------
// Host launcher
// ---------------------------------------------------------------------------
extern "C" void kernel_launch(void* const* d_in, const int* in_sizes, int n_in,
                              void* d_out, int out_size, void* d_ws, size_t ws_size,
                              hipStream_t stream) {
  (void)in_sizes; (void)n_in; (void)out_size; (void)ws_size;
  const float* x     = (const float*)d_in[0];
  const float* Wqkv  = (const float*)d_in[1];
  const float* bqkv  = (const float*)d_in[2];
  const float* Wproj = (const float*)d_in[3];
  const float* bproj = (const float*)d_in[4];
  float* out = (float*)d_out;

  char* ws = (char*)d_ws;
  size_t off = 0;
  auto carve = [&](size_t bytes) -> char* {
    char* p = ws + off;
    off += (bytes + 255) & ~(size_t)255;
    return p;
  };
  bf16* xb     = (bf16*)carve((size_t)8388608 * 2);
  bf16* WqkvT  = (bf16*)carve((size_t)3072 * 1024 * 2);
  bf16* WprojT = (bf16*)carve((size_t)1024 * 1024 * 2);
  bf16* Qb     = (bf16*)carve((size_t)8388608 * 2);
  bf16* Kb     = (bf16*)carve((size_t)8388608 * 2);
  bf16* Vb     = (bf16*)carve((size_t)8388608 * 2);
  bf16* Yb     = (bf16*)carve((size_t)8388608 * 2);

  k_cvt4<<<8192, 256, 0, stream>>>(x, xb, 8388608 / 4);
  k_cvt_transpose<<<12288, 256, 0, stream>>>(Wqkv, WqkvT, 1024, 3072);
  k_cvt_transpose<<<4096, 256, 0, stream>>>(Wproj, WprojT, 1024, 1024);
  k_qkv_gemm<<<3072, 256, 0, stream>>>(xb, WqkvT, bqkv, Qb, Kb, Vb);
  k_attn<<<1024, 256, 0, stream>>>(Qb, Kb, Vb, Yb);
  k_proj_gemm<<<1024, 256, 0, stream>>>(Yb, WprojT, bproj, out);
}
